// Extractor_43387759624796
// MI455X (gfx1250) — matrix operand order repaired and checked by reading, not compiled, not run on hardware
//
#include <hip/hip_runtime.h>
#include <hip/hip_bf16.h>
#include <stdint.h>

typedef unsigned short u16;
typedef unsigned int   u32;
typedef __attribute__((ext_vector_type(16))) __bf16 v16bf;
typedef __attribute__((ext_vector_type(8)))  float  v8f;

union FragU {
    uint4 q[2];
    v16bf v;
};

__device__ __forceinline__ u16 f2bf(float f) {
    u32 u = __float_as_uint(f);
    u32 r = (u + 0x7FFFu + ((u >> 16) & 1u)) >> 16;   // round-to-nearest-even
    return (u16)r;
}
__device__ __forceinline__ float bf2f(u16 h) {
    return __uint_as_float(((u32)h) << 16);
}

// ---------------------------------------------------------------- utilities
__global__ void zero_f32_kernel(float* __restrict__ p, int n) {
    int i = blockIdx.x * blockDim.x + threadIdx.x;
    if (i < n) p[i] = 0.0f;
}
__global__ void zero_u16_kernel(u16* __restrict__ p, int n) {
    int i = blockIdx.x * blockDim.x + threadIdx.x;
    if (i < n) p[i] = 0;
}

// fp32 -> bf16 with optional per-row mean scaling (for "mean" aggregation)
__global__ void cvt_bf16_kernel(const float* __restrict__ in,
                                const float* __restrict__ deg,
                                u16* __restrict__ out, int rows, int F) {
    int i = blockIdx.x * blockDim.x + threadIdx.x;
    if (i >= rows * F) return;
    float v = in[i];
    if (deg) {
        float d = deg[i / F];
        v *= 1.0f / fmaxf(d, 1.0f);
    }
    out[i] = f2bf(v);
}

// in-degree of destination nodes
__global__ void deg_kernel(const int* __restrict__ ei, int nE, float* __restrict__ deg) {
    int e = blockIdx.x * blockDim.x + threadIdx.x;
    if (e < nE) atomicAdd(deg + ei[nE + e], 1.0f);
}

// msg[dst] += x[src]  (bf16 source, fp32 atomic accumulate; 8 features/thread)
__global__ void scatter_kernel(const u16* __restrict__ xbf, const int* __restrict__ ei,
                               int nE, int F, float* __restrict__ msg) {
    int t   = blockIdx.x * blockDim.x + threadIdx.x;
    int lpe = F >> 3;                       // lanes per edge (8 feats each)
    int e   = t / lpe;
    if (e >= nE) return;
    int c   = (t - e * lpe) << 3;
    int src = ei[e];
    int dst = ei[nE + e];
    uint4 q = *(const uint4*)(xbf + (size_t)src * F + c);
    const u16* us = (const u16*)&q;
    float* mp = msg + (size_t)dst * F + c;
#pragma unroll
    for (int i = 0; i < 8; ++i) atomicAdd(mp + i, bf2f(us[i]));
}

// Pack fp32 weight W[K,256] into WMMA B-fragment order (bf16), kt-major:
//   frag = kt*16 + nt ; lane holds col nt*16+(lane&15), k = kt*32 + (lane>>4)*8 + (e&7) + (e>>3)*16
//   out[(frag*32 + lane)*16 + e]
// => advancing kt is a flat +16KB stride; the 4 nt-fragments a wave needs sit
//    at fixed 1KB offsets (fits the 24-bit vmem immediate).
__global__ void pack_w_kernel(const float* __restrict__ W, u16* __restrict__ out, int K) {
    int t  = blockIdx.x * blockDim.x + threadIdx.x;
    int KT = K >> 5;
    if (t >= KT * 16 * 32) return;
    int lane = t & 31;
    int frag = t >> 5;
    int nt = frag & 15;
    int kt = frag >> 4;
    const float* wp = W + (size_t)(kt * 32) * 256 + nt * 16 + (lane & 15);
    u16* op = out + (size_t)t * 16;
#pragma unroll
    for (int e = 0; e < 16; ++e) op[e] = f2bf(wp[(size_t)((lane >> 4) * 8 + (e & 7) + (e >> 3) * 16) * 256]);
}

// ------------------------------------------------------------------- GEMM
// out[M,256] = Ax@Wx + Am@Wm + bias  (optional ReLU)
// Block: 256 threads = 8 waves stacked in M; block tile 256 rows x 64 cols.
// Wave tile 32x64: 2 M-subtiles x 4 N-tiles = 8 WMMA accumulators; the two
// M-subtiles share each B fragment (8 WMMAs per 12 b128 loads).
__global__ void __launch_bounds__(256, 1)
gemm_wmma_kernel(const u16* __restrict__ Ax, const u16* __restrict__ Am,
                 const u16* __restrict__ Wx, const u16* __restrict__ Wm,
                 const float* __restrict__ bias,
                 u16* __restrict__ outbf, float* __restrict__ outf,
                 int M, int K, int relu) {
    const int lane = threadIdx.x & 31;
    const int wave = threadIdx.x >> 5;
    const int row0 = blockIdx.x * 256 + wave * 32;
    const int col0 = blockIdx.y * 64;
    const int KT   = K >> 5;

    v8f acc[2][4];
#pragma unroll
    for (int mt = 0; mt < 2; ++mt)
#pragma unroll
        for (int nt = 0; nt < 4; ++nt)
#pragma unroll
            for (int r = 0; r < 8; ++r) acc[mt][nt][r] = 0.0f;

    // 16-bit A fragment: lane<16 -> K {0..7,16..23}, lane>=16 -> K {8..15,24..31}
    const int kbase = (lane >> 4) * 8;
    const int arow  = row0 + (lane & 15);

    for (int s = 0; s < 2; ++s) {
        const u16* A  = s ? Am : Ax;
        const u16* Wp = s ? Wm : Wx;
        if (!A) continue;
        // A row pointers for the two M-subtiles (fixed 16*K element gap)
        const u16* ar0 = A + (size_t)arow * K + kbase;
        const u16* ar1 = ar0 + (size_t)16 * K;
        // B base: kt-major slabs of 16KB; nt offsets are 1KB immediates
        const u16* bbase = Wp + ((size_t)blockIdx.y * 4 * 32 + lane) * 16;
        for (int kt = 0; kt < KT; ++kt) {
            FragU a0, a1;
            const u16* ap0 = ar0 + kt * 32;
            const u16* ap1 = ar1 + kt * 32;
            a0.q[0] = *(const uint4*)(ap0);
            a0.q[1] = *(const uint4*)(ap0 + 16);
            a1.q[0] = *(const uint4*)(ap1);
            a1.q[1] = *(const uint4*)(ap1 + 16);
            const u16* bp = bbase + (size_t)kt * (16 * 32 * 16);
#pragma unroll
            for (int nt = 0; nt < 4; ++nt) {
                FragU b;
                b.q[0] = *(const uint4*)(bp + nt * 512);
                b.q[1] = *(const uint4*)(bp + nt * 512 + 8);
                acc[0][nt] = __builtin_amdgcn_wmma_f32_16x16x32_bf16(
                    false, a0.v, false, b.v, (short)0, acc[0][nt], false, false);
                acc[1][nt] = __builtin_amdgcn_wmma_f32_16x16x32_bf16(
                    false, a1.v, false, b.v, (short)0, acc[1][nt], false, false);
            }
        }
    }

    // Epilogue.  C layout: VGPR r, lanes 0-15 -> (M=r, N=lane), lanes 16-31 -> (M=8+r)
    const int nlo   = lane & 15;
    const int mhalf = (lane >> 4) * 8;
#pragma unroll
    for (int mt = 0; mt < 2; ++mt) {
#pragma unroll
        for (int nt = 0; nt < 4; ++nt) {
            int col  = col0 + nt * 16 + nlo;
            float bb = bias ? bias[col] : 0.0f;
#pragma unroll
            for (int r = 0; r < 8; ++r) {
                int row = row0 + mt * 16 + mhalf + r;
                if (row < M) {
                    float v = acc[mt][nt][r] + bb;
                    if (relu) v = fmaxf(v, 0.0f);
                    size_t o = (size_t)row * 256 + col;
                    if (outbf) outbf[o] = f2bf(v);
                    if (outf)  outf[o]  = v;
                }
            }
        }
    }
}

// ------------------------------------------------------------------- host
extern "C" void kernel_launch(void* const* d_in, const int* in_sizes, int n_in,
                              void* d_out, int out_size, void* d_ws, size_t ws_size,
                              hipStream_t stream) {
    const float* x      = (const float*)d_in[0];
    const int*   ei_con = (const int*)d_in[1];
    const int*   ei_dst = (const int*)d_in[2];
    const int*   ei_trn = (const int*)d_in[3];
    const int nCon = in_sizes[1] / 2;
    const int nDst = in_sizes[2] / 2;
    const int nTrn = in_sizes[3] / 2;

    const float *Wroot[5], *Wrel[5], *bias[5];
    for (int i = 0; i < 5; ++i) {
        Wroot[i] = (const float*)d_in[4 + 3 * i];
        Wrel[i]  = (const float*)d_in[5 + 3 * i];
        bias[i]  = (const float*)d_in[6 + 3 * i];
    }
    const float* WL = (const float*)d_in[19];  // [2,256,256]
    const float* bL = (const float*)d_in[20];  // [2,256]

    const int M = 10000, Mpad = 10240, H = 256, INF = 128;

    char* ws = (char*)d_ws;
    size_t off = 0;
    auto alloc = [&](size_t bytes) -> char* {
        char* p = ws + off;
        off += (bytes + 255) & ~(size_t)255;
        return p;
    };
    u16*   xA    = (u16*)alloc((size_t)Mpad * H * 2);
    u16*   xB    = (u16*)alloc((size_t)Mpad * H * 2);
    float* msg   = (float*)alloc((size_t)Mpad * H * 4);
    u16*   msgbf = (u16*)alloc((size_t)Mpad * H * 2);
    float* deg   = (float*)alloc((size_t)Mpad * 4);
    u16* pkRoot[5], *pkRel[5], *pkL[2];
    for (int i = 0; i < 5; ++i) {
        int K = (i == 0) ? INF : H;
        pkRoot[i] = (u16*)alloc((size_t)K * 256 * 2);
        pkRel[i]  = (u16*)alloc((size_t)K * 256 * 2);
    }
    for (int i = 0; i < 2; ++i) pkL[i] = (u16*)alloc((size_t)H * 256 * 2);
    (void)ws_size; (void)n_in; (void)out_size;

    auto cdiv = [](int a, int b) { return (a + b - 1) / b; };

    // Pack all weights into WMMA fragment order (bf16, kt-major)
    for (int i = 0; i < 5; ++i) {
        int K = (i == 0) ? INF : H;
        int n = (K >> 5) * 16 * 32;
        pack_w_kernel<<<cdiv(n, 256), 256, 0, stream>>>(Wroot[i], pkRoot[i], K);
        pack_w_kernel<<<cdiv(n, 256), 256, 0, stream>>>(Wrel[i], pkRel[i], K);
    }
    {
        int n = (H >> 5) * 16 * 32;
        pack_w_kernel<<<cdiv(n, 256), 256, 0, stream>>>(WL, pkL[0], H);
        pack_w_kernel<<<cdiv(n, 256), 256, 0, stream>>>(WL + 256 * 256, pkL[1], H);
    }

    // Zero feature buffers (defines padded rows), convert input x -> bf16
    zero_u16_kernel<<<cdiv(Mpad * H, 256), 256, 0, stream>>>(xA, Mpad * H);
    zero_u16_kernel<<<cdiv(Mpad * H, 256), 256, 0, stream>>>(xB, Mpad * H);
    cvt_bf16_kernel<<<cdiv(M * INF, 256), 256, 0, stream>>>(x, nullptr, xA, M, INF);

    // Degree of connections graph (for conv3 "mean")
    zero_f32_kernel<<<cdiv(Mpad, 256), 256, 0, stream>>>(deg, Mpad);
    deg_kernel<<<cdiv(nCon, 256), 256, 0, stream>>>(ei_con, nCon, deg);

    dim3 ggrid((unsigned)cdiv(M, 256), 4);

    auto gconv = [&](const u16* xin, int K, const int* ei, int nE,
                     u16* pk_root, u16* pk_rel, const float* b,
                     const float* degp, int relu, u16* outbf, float* outf) {
        int nMsg = Mpad * K;
        zero_f32_kernel<<<cdiv(nMsg, 256), 256, 0, stream>>>(msg, nMsg);
        int nThr = nE * (K >> 3);
        scatter_kernel<<<cdiv(nThr, 256), 256, 0, stream>>>(xin, ei, nE, K, msg);
        cvt_bf16_kernel<<<cdiv(nMsg, 256), 256, 0, stream>>>(msg, degp, msgbf, Mpad, K);
        gemm_wmma_kernel<<<ggrid, 256, 0, stream>>>(xin, msgbf, pk_root, pk_rel, b,
                                                    outbf, outf, M, K, relu);
    };
    auto lin = [&](const u16* xin, u16* pkW, const float* b, int relu,
                   u16* outbf, float* outf) {
        gemm_wmma_kernel<<<ggrid, 256, 0, stream>>>(xin, nullptr, pkW, nullptr, b,
                                                    outbf, outf, M, H, relu);
    };

    // conv1: connections, add, relu          (K = 128)
    gconv(xA, INF, ei_con, nCon, pkRoot[0], pkRel[0], bias[0], nullptr, 1, xB, nullptr);
    // conv2: trains, add, no activation
    gconv(xB, H, ei_trn, nTrn, pkRoot[1], pkRel[1], bias[1], nullptr, 0, xA, nullptr);
    // conv3 x2: connections, mean, relu
    gconv(xA, H, ei_con, nCon, pkRoot[2], pkRel[2], bias[2], deg, 1, xB, nullptr);
    gconv(xB, H, ei_con, nCon, pkRoot[2], pkRel[2], bias[2], deg, 1, xA, nullptr);
    // conv4: destinations, add, relu
    gconv(xA, H, ei_dst, nDst, pkRoot[3], pkRel[3], bias[3], nullptr, 1, xB, nullptr);
    // conv5 x2: connections, add, relu
    gconv(xB, H, ei_con, nCon, pkRoot[4], pkRel[4], bias[4], nullptr, 1, xA, nullptr);
    gconv(xA, H, ei_con, nCon, pkRoot[4], pkRel[4], bias[4], nullptr, 1, xB, nullptr);
    // two final linears (no activation); last one writes fp32 to d_out
    lin(xB, pkL[0], bL, 0, xA, nullptr);
    lin(xA, pkL[1], bL + 256, 0, nullptr, (float*)d_out);
}
